// Switch_39144331936231
// MI455X (gfx1250) — matrix operand order repaired and checked by reading, not compiled, not run on hardware
//
#include <hip/hip_runtime.h>

// ---------------------------------------------------------------------------
// Problem constants (from reference: B=4, T=2048, D=1024, E=8)
// ---------------------------------------------------------------------------
#define N_TOK 8192      // B*T
#define D_DIM 1024
#define E_NUM 8
#define CAP   1024      // expert capacity = N/E

typedef __attribute__((ext_vector_type(16))) _Float16 v16h;
typedef __attribute__((ext_vector_type(8)))  _Float16 v8h;
typedef __attribute__((ext_vector_type(4)))  _Float16 v4h;
typedef __attribute__((ext_vector_type(8)))  float    v8f;

// ---------------------------------------------------------------------------
// CDNA5 async global->LDS copy (ASYNCcnt-tracked, no VGPR round trip).
// LDS byte offset = low 32 bits of the generic pointer (ISA 10.2: the LDS
// aperture is identified purely by addr[63:32]; addr[31:0] is the DS address).
// ---------------------------------------------------------------------------
__device__ __forceinline__ void async_load_b128(void* lds_dst, const void* gsrc) {
    unsigned lds_addr = (unsigned)(size_t)lds_dst;
    asm volatile("global_load_async_to_lds_b128 %0, %1, off"
                 :: "v"(lds_addr), "v"(gsrc)
                 : "memory");
}
__device__ __forceinline__ void wait_async_le8() {
    asm volatile("s_wait_asynccnt 0x8" ::: "memory");
}
__device__ __forceinline__ void wait_async_0() {
    asm volatile("s_wait_asynccnt 0x0" ::: "memory");
}

// ---------------------------------------------------------------------------
// 0) Init: slotmap = -1, out = 0 (dropped tokens / slot 0 stay zero)
// ---------------------------------------------------------------------------
__global__ void moe_init_kernel(int* __restrict__ slotmap, float* __restrict__ out) {
    int i = blockIdx.x * blockDim.x + threadIdx.x;
    if (i < E_NUM * CAP) slotmap[i] = -1;
    const int total4 = (N_TOK * D_DIM) / 4;   // 2M float4
    float4 z; z.x = 0.f; z.y = 0.f; z.z = 0.f; z.w = 0.f;
    for (int j = i; j < total4; j += gridDim.x * blockDim.x)
        ((float4*)out)[j] = z;
}

// ---------------------------------------------------------------------------
// 1) Router: one wave per token. logits = x@Wr + br ; gate = 1/sum(exp(l-lmax))
// ---------------------------------------------------------------------------
__global__ __launch_bounds__(256) void moe_router_kernel(
    const float* __restrict__ x, const float* __restrict__ Wr,
    const float* __restrict__ br, float* __restrict__ gate, int* __restrict__ eidx) {
    const int lane = threadIdx.x & 31;
    const int wave = threadIdx.x >> 5;
    const int n = blockIdx.x * 8 + wave;

    float acc[E_NUM];
#pragma unroll
    for (int e = 0; e < E_NUM; ++e) acc[e] = 0.f;

    const float* xr = x + (long)n * D_DIM;
    for (int d = lane; d < D_DIM; d += 32) {
        const float xv = xr[d];
        const float* wr = Wr + d * E_NUM;
#pragma unroll
        for (int e = 0; e < E_NUM; ++e) acc[e] += xv * wr[e];
    }
#pragma unroll
    for (int off = 16; off > 0; off >>= 1)
#pragma unroll
        for (int e = 0; e < E_NUM; ++e) acc[e] += __shfl_down(acc[e], off, 32);

    if (lane == 0) {
#pragma unroll
        for (int e = 0; e < E_NUM; ++e) acc[e] += br[e];
        float m = acc[0]; int am = 0;
#pragma unroll
        for (int e = 1; e < E_NUM; ++e) { if (acc[e] > m) { m = acc[e]; am = e; } }
        float s = 0.f;
#pragma unroll
        for (int e = 0; e < E_NUM; ++e) s += __expf(acc[e] - m);
        gate[n] = 1.0f / s;     // max softmax prob
        eidx[n] = am;
    }
}

// ---------------------------------------------------------------------------
// 2) Position scan: 1 block, wave w owns expert w. Ballot prefix over chunks
//    of 32 tokens reproduces the serial cumsum exactly. slot = 1-based pos,
//    kept iff pos < CAP (slot 0 never used, matching the reference quirk).
// ---------------------------------------------------------------------------
__global__ __launch_bounds__(256) void moe_scan_kernel(
    const int* __restrict__ eidx, int* __restrict__ slotmap) {
    const int wave = threadIdx.x >> 5;   // expert id (8 waves)
    const int lane = threadIdx.x & 31;
    int run = 0;
    for (int base = 0; base < N_TOK; base += 32) {
        const int n = base + lane;
        const int e = eidx[n];
        const unsigned long long m = __ballot(e == wave);
        const int prefix = __popcll(m & ((1ull << lane) - 1ull));
        const int p = run + prefix + 1;          // 1-based position
        if (e == wave && p < CAP)
            slotmap[wave * CAP + p] = n;
        run += __popcll(m);
    }
}

// ---------------------------------------------------------------------------
// 3) Weight convert fp32 -> f16, transposed: dst[e][c][r] = src[e][r][c].
// ---------------------------------------------------------------------------
__global__ __launch_bounds__(256) void moe_convert_kernel(
    const float* __restrict__ W1, const float* __restrict__ W2,
    _Float16* __restrict__ W1t, _Float16* __restrict__ W2t) {
    const float* src = blockIdx.z ? W2 : W1;
    _Float16*   dst = blockIdx.z ? W2t : W1t;
    const int e  = blockIdx.y;
    const int td = (blockIdx.x >> 5) * 32;   // src row tile
    const int tf = (blockIdx.x & 31) * 32;   // src col tile

    __shared__ float tile[32][33];
    const int t4 = threadIdx.x * 4;
    const int r = t4 >> 5, c = t4 & 31;
    const float4 v = *(const float4*)&src[((size_t)e * D_DIM + td + r) * D_DIM + tf + c];
    tile[r][c + 0] = v.x; tile[r][c + 1] = v.y; tile[r][c + 2] = v.z; tile[r][c + 3] = v.w;
    __syncthreads();

    v4h o;
#pragma unroll
    for (int i = 0; i < 4; ++i) o[i] = (_Float16)tile[c + i][r];
    *(v4h*)&dst[((size_t)e * D_DIM + tf + r) * D_DIM + td + c] = o;
}

// ---------------------------------------------------------------------------
// 4) Dispatch: Xe[e*CAP + slot, :] = gate[tok] * x[tok, :]  (f16), zeros else
// ---------------------------------------------------------------------------
__global__ __launch_bounds__(256) void moe_dispatch_kernel(
    const float* __restrict__ x, const float* __restrict__ gate,
    const int* __restrict__ slotmap, _Float16* __restrict__ Xe) {
    const int row = blockIdx.x;                 // e*CAP + slot
    const int tok = slotmap[row];
    const int t4 = threadIdx.x * 4;
    v4h o;
    if (tok < 0) {
        o[0] = (_Float16)0.f; o[1] = (_Float16)0.f; o[2] = (_Float16)0.f; o[3] = (_Float16)0.f;
    } else {
        const float w = gate[tok];
        const float4 xv = *(const float4*)&x[(size_t)tok * D_DIM + t4];
        o[0] = (_Float16)(w * xv.x); o[1] = (_Float16)(w * xv.y);
        o[2] = (_Float16)(w * xv.z); o[3] = (_Float16)(w * xv.w);
    }
    *(v4h*)&Xe[(size_t)row * D_DIM + t4] = o;
}

// ---------------------------------------------------------------------------
// WMMA fragment helpers (ISA 7.12.2 layouts, wave32)
//   A 16x32 f16 : lane half h, row lr ; elems i<8 -> K = h*8+i, i>=8 -> K = 16+h*8+(i-8)
//   B 32x16 f16 : lane half h, col lr ; elems i<8 -> K = h*8+i, i>=8 -> K = 16+h*8+(i-8)  (tile stored [N][K])
// LDS row stride = 72 halves (144 B): BK=64 tile, every 8-half chunk 16 B aligned.
// ---------------------------------------------------------------------------
#define BK 64
#define LDSS 72

__device__ __forceinline__ v16h frag_a(const _Float16* base, int row, int kk, int half) {
    const _Float16* p = base + row * LDSS + kk;
    v8h lo = *(const v8h*)(p + half * 8);
    v8h hi = *(const v8h*)(p + 16 + half * 8);
    return __builtin_shufflevector(lo, hi, 0,1,2,3,4,5,6,7,8,9,10,11,12,13,14,15);
}
__device__ __forceinline__ v16h frag_b(const _Float16* base, int col, int kk, int half) {
    const _Float16* p = base + col * LDSS + kk + half * 8;
    v8h lo = *(const v8h*)(p);
    v8h hi = *(const v8h*)(p + 16);
    return __builtin_shufflevector(lo, hi, 0,1,2,3,4,5,6,7,8,9,10,11,12,13,14,15);
}

// Issue one 128x64 f16 A-tile + 128x64 f16 B-tile as async copies.
// 1024 16B segments per tile, 256 threads -> 4 A + 4 B = 8 async ops/thread.
__device__ __forceinline__ void issue_tile_async(
    _Float16* AsBuf, _Float16* BsBuf,
    const _Float16* __restrict__ Ag, const _Float16* __restrict__ Bg,
    int k0, int t) {
#pragma unroll
    for (int s = t; s < 1024; s += 256) {
        const int row = s >> 3, c8 = (s & 7) << 3;
        async_load_b128(AsBuf + row * LDSS + c8, Ag + (size_t)row * D_DIM + k0 + c8);
        async_load_b128(BsBuf + row * LDSS + c8, Bg + (size_t)row * D_DIM + k0 + c8);
    }
}

// ---------------------------------------------------------------------------
// 5) GEMM1: H[e] = relu(Xe[e] @ W1[e] + b1[e])  (f16 out, f32 accumulate)
//    256 thr (8 waves), block tile 128x128, BK=64, double-buffered async LDS.
// ---------------------------------------------------------------------------
__global__ __launch_bounds__(256) void moe_gemm1_kernel(
    const _Float16* __restrict__ Xe, const _Float16* __restrict__ W1t,
    const float* __restrict__ b1, _Float16* __restrict__ H) {
    const int e  = blockIdx.z;
    const int m0 = blockIdx.y * 128;
    const int n0 = blockIdx.x * 128;

    __shared__ __align__(16) _Float16 As[2][128 * LDSS];
    __shared__ __align__(16) _Float16 Bs[2][128 * LDSS];

    const int t = threadIdx.x, lane = t & 31, wave = t >> 5;
    const int wm = wave >> 1, wn = wave & 1;
    const int half = lane >> 4, lr = lane & 15;

    const _Float16* Ag = Xe  + ((size_t)e * CAP   + m0) * D_DIM;
    const _Float16* Bg = W1t + ((size_t)e * D_DIM + n0) * D_DIM;

    v8f acc[2][4];
#pragma unroll
    for (int mi = 0; mi < 2; ++mi)
#pragma unroll
        for (int ni = 0; ni < 4; ++ni)
#pragma unroll
            for (int j = 0; j < 8; ++j) acc[mi][ni][j] = 0.f;

    issue_tile_async(As[0], Bs[0], Ag, Bg, 0, t);
    int cur = 0;

    for (int k0 = 0; k0 < D_DIM; k0 += BK) {
        __syncthreads();                       // prior reads of buf cur^1 done
        if (k0 + BK < D_DIM) {
            issue_tile_async(As[cur ^ 1], Bs[cur ^ 1], Ag, Bg, k0 + BK, t);
            wait_async_le8();                  // oldest 8 (current buffer) done
        } else {
            wait_async_0();
        }
        __syncthreads();                       // current buffer visible to all

        const _Float16* Ab = As[cur];
        const _Float16* Bb = Bs[cur];
#pragma unroll
        for (int kk = 0; kk < BK; kk += 32) {
            v16h a[2], b[4];
#pragma unroll
            for (int mi = 0; mi < 2; ++mi) a[mi] = frag_a(Ab, wm * 32 + mi * 16 + lr, kk, half);
#pragma unroll
            for (int ni = 0; ni < 4; ++ni) b[ni] = frag_b(Bb, wn * 64 + ni * 16 + lr, kk, half);
#pragma unroll
            for (int mi = 0; mi < 2; ++mi)
#pragma unroll
                for (int ni = 0; ni < 4; ++ni)
                    acc[mi][ni] = __builtin_amdgcn_wmma_f32_16x16x32_f16(
                        false, a[mi], false, b[ni], (short)0, acc[mi][ni], false, false);
        }
        cur ^= 1;
    }

    // Epilogue: bias + relu, store f16. C/D layout: lane col = lr, rows = half*8 + j.
    _Float16* Hb = H + (size_t)e * CAP * D_DIM;
#pragma unroll
    for (int ni = 0; ni < 4; ++ni) {
        const int coln = n0 + wn * 64 + ni * 16 + lr;
        const float bv = b1[e * D_DIM + coln];
#pragma unroll
        for (int mi = 0; mi < 2; ++mi) {
            const int rbase = m0 + wm * 32 + mi * 16 + half * 8;
#pragma unroll
            for (int j = 0; j < 8; ++j) {
                float hv = acc[mi][ni][j] + bv;
                hv = fmaxf(hv, 0.f);
                Hb[(size_t)(rbase + j) * D_DIM + coln] = (_Float16)hv;
            }
        }
    }
}

// ---------------------------------------------------------------------------
// 6) GEMM2 + combine: out[tok] = gate[tok] * (H[e] @ W2[e] + b2[e])[slot]
// ---------------------------------------------------------------------------
__global__ __launch_bounds__(256) void moe_gemm2_kernel(
    const _Float16* __restrict__ H, const _Float16* __restrict__ W2t,
    const float* __restrict__ b2, const int* __restrict__ slotmap,
    const float* __restrict__ gate, float* __restrict__ out) {
    const int e  = blockIdx.z;
    const int m0 = blockIdx.y * 128;
    const int n0 = blockIdx.x * 128;

    __shared__ __align__(16) _Float16 As[2][128 * LDSS];
    __shared__ __align__(16) _Float16 Bs[2][128 * LDSS];
    __shared__ int   s_tok[128];
    __shared__ float s_w[128];

    const int t = threadIdx.x, lane = t & 31, wave = t >> 5;
    const int wm = wave >> 1, wn = wave & 1;
    const int half = lane >> 4, lr = lane & 15;

    if (t < 128) {
        const int tok = slotmap[e * CAP + m0 + t];
        s_tok[t] = tok;
        s_w[t] = (tok >= 0) ? gate[tok] : 0.f;
    }

    const _Float16* Ag = H   + ((size_t)e * CAP   + m0) * D_DIM;
    const _Float16* Bg = W2t + ((size_t)e * D_DIM + n0) * D_DIM;

    v8f acc[2][4];
#pragma unroll
    for (int mi = 0; mi < 2; ++mi)
#pragma unroll
        for (int ni = 0; ni < 4; ++ni)
#pragma unroll
            for (int j = 0; j < 8; ++j) acc[mi][ni][j] = 0.f;

    issue_tile_async(As[0], Bs[0], Ag, Bg, 0, t);
    int cur = 0;

    for (int k0 = 0; k0 < D_DIM; k0 += BK) {
        __syncthreads();
        if (k0 + BK < D_DIM) {
            issue_tile_async(As[cur ^ 1], Bs[cur ^ 1], Ag, Bg, k0 + BK, t);
            wait_async_le8();
        } else {
            wait_async_0();
        }
        __syncthreads();

        const _Float16* Ab = As[cur];
        const _Float16* Bb = Bs[cur];
#pragma unroll
        for (int kk = 0; kk < BK; kk += 32) {
            v16h a[2], b[4];
#pragma unroll
            for (int mi = 0; mi < 2; ++mi) a[mi] = frag_a(Ab, wm * 32 + mi * 16 + lr, kk, half);
#pragma unroll
            for (int ni = 0; ni < 4; ++ni) b[ni] = frag_b(Bb, wn * 64 + ni * 16 + lr, kk, half);
#pragma unroll
            for (int mi = 0; mi < 2; ++mi)
#pragma unroll
                for (int ni = 0; ni < 4; ++ni)
                    acc[mi][ni] = __builtin_amdgcn_wmma_f32_16x16x32_f16(
                        false, a[mi], false, b[ni], (short)0, acc[mi][ni], false, false);
        }
        cur ^= 1;
    }

    // Epilogue: bias + gate, scatter rows back to their tokens.
#pragma unroll
    for (int ni = 0; ni < 4; ++ni) {
        const int d = n0 + wn * 64 + ni * 16 + lr;
        const float bv = b2[e * D_DIM + d];
#pragma unroll
        for (int mi = 0; mi < 2; ++mi) {
            const int rbase = wm * 32 + mi * 16 + half * 8;   // local row in block
#pragma unroll
            for (int j = 0; j < 8; ++j) {
                const int lrow = rbase + j;
                const int tok = s_tok[lrow];
                if (tok >= 0)
                    out[(size_t)tok * D_DIM + d] = s_w[lrow] * (acc[mi][ni][j] + bv);
            }
        }
    }
}

// ---------------------------------------------------------------------------
// Host launcher
// ---------------------------------------------------------------------------
extern "C" void kernel_launch(void* const* d_in, const int* in_sizes, int n_in,
                              void* d_out, int out_size, void* d_ws, size_t ws_size,
                              hipStream_t stream) {
    const float* x  = (const float*)d_in[0];   // [N, D]
    const float* Wr = (const float*)d_in[1];   // [D, E]
    const float* br = (const float*)d_in[2];   // [E]
    const float* W1 = (const float*)d_in[3];   // [E, D, D]
    const float* b1 = (const float*)d_in[4];   // [E, D]
    const float* W2 = (const float*)d_in[5];   // [E, D, D]
    const float* b2 = (const float*)d_in[6];   // [E, D]
    float* out = (float*)d_out;                // [N, D]

    // Workspace carve-up (256 B aligned)
    char* ws = (char*)d_ws;
    size_t off = 0;
    auto carve = [&](size_t bytes) {
        char* p = ws + off;
        off = (off + bytes + 255) & ~(size_t)255;
        return p;
    };
    float*    gate    = (float*)   carve(N_TOK * sizeof(float));
    int*      eidx    = (int*)     carve(N_TOK * sizeof(int));
    int*      slotmap = (int*)     carve(E_NUM * CAP * sizeof(int));
    _Float16* Xe      = (_Float16*)carve((size_t)E_NUM * CAP * D_DIM * sizeof(_Float16));
    _Float16* W1t     = (_Float16*)carve((size_t)E_NUM * D_DIM * D_DIM * sizeof(_Float16));
    _Float16* W2t     = (_Float16*)carve((size_t)E_NUM * D_DIM * D_DIM * sizeof(_Float16));
    _Float16* Hb      = (_Float16*)carve((size_t)E_NUM * CAP * D_DIM * sizeof(_Float16));
    (void)ws_size; (void)in_sizes; (void)n_in; (void)out_size;

    moe_init_kernel<<<2048, 256, 0, stream>>>(slotmap, out);
    moe_router_kernel<<<N_TOK / 8, 256, 0, stream>>>(x, Wr, br, gate, eidx);
    moe_scan_kernel<<<1, 256, 0, stream>>>(eidx, slotmap);
    moe_convert_kernel<<<dim3(1024, E_NUM, 2), 256, 0, stream>>>(W1, W2, W1t, W2t);
    moe_dispatch_kernel<<<E_NUM * CAP, 256, 0, stream>>>(x, gate, slotmap, Xe);
    moe_gemm1_kernel<<<dim3(8, 8, E_NUM), 256, 0, stream>>>(Xe, W1t, b1, Hb);
    moe_gemm2_kernel<<<dim3(8, 8, E_NUM), 256, 0, stream>>>(Hb, W2t, b2, slotmap, gate, out);
}
